// SelfAttention_635655160086
// MI455X (gfx1250) — matrix operand order repaired and checked by reading, not compiled, not run on hardware
//
#include <hip/hip_runtime.h>

// ---------------------------------------------------------------------------
// Problem constants (reference: B=2, S=2048, D=1024, H=16, HD=64)
// ---------------------------------------------------------------------------
#define BATCH 2
#define SEQ   2048
#define DIM   1024
#define NHEAD 16
#define HDIM  64
#define MROWS (BATCH * SEQ)      // 4096 rows for the dense GEMMs
#define SOFTMAX_SCALE 0.125f     // 1/sqrt(64)

typedef __attribute__((ext_vector_type(16))) __bf16 v16bf;
typedef __attribute__((ext_vector_type(8)))  __bf16 v8bf;
typedef __attribute__((ext_vector_type(8)))  float  v8f;
typedef __attribute__((ext_vector_type(4)))  unsigned int u32x4;
typedef __attribute__((ext_vector_type(8)))  int          i32x8;
typedef __attribute__((ext_vector_type(4)))  int          i32x4;

union V16U { v16bf v; v8bf h[2]; };

static __device__ inline v8f vzero8() {
    v8f z;
#pragma unroll
    for (int i = 0; i < 8; ++i) z[i] = 0.0f;
    return z;
}

// A-operand tile (16x32 bf16) from a row-major [rows][stride] bf16 buffer.
// lane: m = lane&15, half = lane>>4. elems 0..7 -> K = half*8+j,
// elems 8..15 -> K = 16 + half*8 + j.
static __device__ inline v16bf load_a_tile(const __bf16* __restrict__ base,
                                           int stride, int m, int half, int k0) {
    V16U u;
    const __bf16* r = base + (size_t)m * stride + k0 + half * 8;
    u.h[0] = *(const v8bf*)(r);
    u.h[1] = *(const v8bf*)(r + 16);
    return u.v;
}

// B-operand tile (32x16 bf16) from an N-major [n][stride] bf16 buffer
// (i.e. B^T stored row-major). lane: n = lane&15, half = lane>>4,
// elems 0..7 -> K = half*8+j, elems 8..15 -> K = 16 + half*8 + j.
static __device__ inline v16bf load_b_tile(const __bf16* __restrict__ base,
                                           int stride, int n, int half, int k0) {
    V16U u;
    const __bf16* r = base + (size_t)n * stride + k0 + half * 8;
    u.h[0] = *(const v8bf*)(r);
    u.h[1] = *(const v8bf*)(r + 16);
    return u.v;
}

// ---------------------------------------------------------------------------
// TDM: 2D tile load global -> LDS with hardware padding.
// Pads 4 DWORDs (8 bf16) after every 32 DWORDs (64 bf16) -> LDS row stride 72.
// D# packing per CDNA5 ISA. Dims in elements.
// This toolchain uses the 6-arg builtin: (g0, g1, g2, g3, g4, cpol).
// ---------------------------------------------------------------------------
static __device__ inline void tdm_load_tile_2d_padded(
        const void* gptr, unsigned lds_byte_addr,
        unsigned tile_d0, unsigned tile_d1,
        unsigned tensor_d0, unsigned tensor_d1,
        unsigned long long stride0_elems) {
    unsigned long long ga = (unsigned long long)gptr;
    u32x4 g0;
    g0[0] = 1u;                                        // count=1, user mode
    g0[1] = lds_byte_addr;                             // lds_addr [63:32]
    g0[2] = (unsigned)(ga & 0xFFFFFFFFu);              // global_addr lo
    g0[3] = (unsigned)((ga >> 32) & 0x01FFFFFFu)       // global_addr hi [56:32]
          | (2u << 30);                                // type = 2 ("image")
    i32x8 g1;
    g1[0] = (int)((1u << 16)                           // data_size = 2 bytes
                | (1u << 20)                           // pad_enable
                | (4u << 22)                           // pad_interval: 32 DW
                | (3u << 25));                         // pad_amount: 4 DW
    g1[1] = (int)((tensor_d0 & 0xFFFFu) << 16);        // tensor_dim0[15:0]
    g1[2] = (int)((tensor_d0 >> 16) | ((tensor_d1 & 0xFFFFu) << 16));
    g1[3] = (int)((tensor_d1 >> 16) | (tile_d0 << 16));
    g1[4] = (int)tile_d1;                              // tile_dim1, tile_dim2=0
    g1[5] = (int)(stride0_elems & 0xFFFFFFFFu);        // tensor_dim0_stride lo
    g1[6] = (int)((stride0_elems >> 32) & 0xFFFFu);    // stride hi, dim1_stride=0
    g1[7] = 0;
    i32x4 z4 = {0, 0, 0, 0};
    i32x8 z8 = {0, 0, 0, 0, 0, 0, 0, 0};
    __builtin_amdgcn_tensor_load_to_lds(g0, g1, z4, z4, z8, 0);
}

// ---------------------------------------------------------------------------
// Kernel 1: weight convert + transpose. w[k][n] f32 -> wt[n][k] bf16
// ---------------------------------------------------------------------------
__global__ __launch_bounds__(256)
void wconv_kernel(const float* __restrict__ w, __bf16* __restrict__ wt) {
    int idx = blockIdx.x * 256 + threadIdx.x;     // over DIM*DIM
    int n = idx >> 10;
    int k = idx & (DIM - 1);
    wt[idx] = (__bf16)w[(size_t)k * DIM + n];
}

// ---------------------------------------------------------------------------
// Kernel 2: tiled WMMA GEMM.  C[M x 1024] = A[M x 1024] * Bt^T + bias
//   Bt is [N][K] bf16 (pre-transposed weights), staged via async->LDS.
//   TA    = float (convert to bf16 on the fly) or __bf16 (direct).
//   MODE  = 0: bf16 output scattered into [B, H, S, HD] head-major layout
//           1: f32 output, plain row-major [M][1024]
// Block: 256 threads (8 waves). BM=128 (16 rows/wave), BN=64, BK=64.
// ---------------------------------------------------------------------------
#define AST 72   // LDS stride (elems): 144B/row = 9*16B, bank-conflict-free
#define BST 72
#define BKK 64

template <typename TA, int MODE>
__global__ __launch_bounds__(256)
void gemm_kernel(const TA* __restrict__ A, const __bf16* __restrict__ Bt,
                 const float* __restrict__ bias, void* __restrict__ Out) {
    __shared__ __bf16 as[128 * AST];
    __shared__ __bf16 bs[64 * BST];

    const int tid  = threadIdx.x;
    const int wave = tid >> 5;
    const int lane = tid & 31;
    const int half = lane >> 4;
    const int ln   = lane & 15;

    const int n0 = blockIdx.x * 64;
    const int m0 = blockIdx.y * 128;

    v8f acc[4];
#pragma unroll
    for (int nt = 0; nt < 4; ++nt) acc[nt] = vzero8();

    for (int kb = 0; kb < DIM; kb += BKK) {
        // ---- B tile (64 n-rows x 64 k) : direct global -> LDS async copy ----
#pragma unroll
        for (int i = 0; i < 2; ++i) {
            int ch  = tid + i * 256;              // 512 chunks of 8 bf16
            int row = ch >> 3;
            int c8  = (ch & 7) * 8;
            unsigned lds = (unsigned)(size_t)&bs[row * BST + c8];
            unsigned long long g =
                (unsigned long long)(Bt + (size_t)(n0 + row) * DIM + kb + c8);
            asm volatile("global_load_async_to_lds_b128 %0, %1, off"
                         :: "v"(lds), "v"(g) : "memory");
        }
        // ---- A tile (128 x 64) into LDS as bf16 ----
        if constexpr (__is_same(TA, float)) {
#pragma unroll
            for (int i = 0; i < 8; ++i) {
                int fi  = tid + i * 256;          // 2048 float4 chunks
                int row = fi >> 4;
                int c4  = (fi & 15) * 4;
                const float4 f = *(const float4*)(A + (size_t)(m0 + row) * DIM + kb + c4);
                __bf16* d = &as[row * AST + c4];
                d[0] = (__bf16)f.x; d[1] = (__bf16)f.y;
                d[2] = (__bf16)f.z; d[3] = (__bf16)f.w;
            }
        } else {
#pragma unroll
            for (int i = 0; i < 4; ++i) {
                int ch  = tid + i * 256;          // 1024 chunks of 8 bf16
                int row = ch >> 3;
                int c8  = (ch & 7) * 8;
                *(v8bf*)(&as[row * AST + c8]) =
                    *(const v8bf*)(A + (size_t)(m0 + row) * DIM + kb + c8);
            }
        }
        // ---- prefetch next k-block into cache while we compute ----
        if (kb + BKK < DIM) {
            __builtin_prefetch(A  + (size_t)(m0 + (tid >> 1)) * DIM + kb + BKK + (tid & 1) * 32, 0, 3);
            __builtin_prefetch(Bt + (size_t)(n0 + (tid >> 2)) * DIM + kb + BKK + (tid & 3) * 16, 0, 3);
        }
        asm volatile("s_wait_asynccnt 0" ::: "memory");
        __syncthreads();

        const __bf16* abase = &as[(wave * 16) * AST];
        v16bf a0 = load_a_tile(abase, AST, ln, half, 0);
        v16bf a1 = load_a_tile(abase, AST, ln, half, 32);
#pragma unroll
        for (int nt = 0; nt < 4; ++nt) {
            v16bf b0 = load_b_tile(bs, BST, nt * 16 + ln, half, 0);
            acc[nt] = __builtin_amdgcn_wmma_f32_16x16x32_bf16(
                false, a0, false, b0, (short)0, acc[nt], false, false);
            v16bf b1 = load_b_tile(bs, BST, nt * 16 + ln, half, 32);
            acc[nt] = __builtin_amdgcn_wmma_f32_16x16x32_bf16(
                false, a1, false, b1, (short)0, acc[nt], false, false);
        }
        __syncthreads();
    }

    // ---- epilogue ----
#pragma unroll
    for (int nt = 0; nt < 4; ++nt) {
#pragma unroll
        for (int v = 0; v < 8; ++v) {
            int row = m0 + wave * 16 + v + 8 * half;
            int col = n0 + nt * 16 + ln;
            float val = acc[nt][v] + bias[col];
            if constexpr (MODE == 0) {
                // scatter into [B, H, S, HD] bf16
                int b = row >> 11;                // / SEQ
                int s = row & (SEQ - 1);
                int h = col >> 6;
                int d = col & (HDIM - 1);
                ((__bf16*)Out)[(((size_t)b * NHEAD + h) * SEQ + s) * HDIM + d] = (__bf16)val;
            } else {
                ((float*)Out)[(size_t)row * DIM + col] = val;
            }
        }
    }
}

// ---------------------------------------------------------------------------
// Kernel 3: flash attention.
//   Q,K,V: bf16 [B, H, S, HD].  O: bf16 [B*S, D] (head-interleaved row-major)
//   Block = 256 threads (8 waves). Each wave owns 16 q rows -> 128 q rows/WG.
//   KV processed in blocks of 64; K tile DMA'd by the Tensor Data Mover with
//   hardware padding producing the stride-72 LDS layout; online softmax f32.
// Grid: (S/128, H, B)
// ---------------------------------------------------------------------------
#define KST 72   // LDS stride: 144B rows -> 16B-aligned, bank-conflict-free

__global__ __launch_bounds__(256)
void attn_kernel(const __bf16* __restrict__ Qg, const __bf16* __restrict__ Kg,
                 const __bf16* __restrict__ Vg, __bf16* __restrict__ Og) {
    __shared__ __bf16 kbuf[64 * KST];          // K block, [kv][hd] (TDM-filled)
    __shared__ __bf16 vbuf[HDIM * KST];        // V block transposed, [hd][kv]
    __shared__ __bf16 pbuf[8 * 16 * KST];      // per-wave P staging

    const int tid  = threadIdx.x;
    const int wave = tid >> 5;
    const int lane = tid & 31;
    const int half = lane >> 4;
    const int ln   = lane & 15;

    const int h  = blockIdx.y;
    const int b  = blockIdx.z;
    const size_t base = ((size_t)b * NHEAD + h) * SEQ * HDIM;
    const int q0 = blockIdx.x * 128 + wave * 16;   // wave's q-row start

    // Q A-tiles for the full HD=64 (two 32-wide k-steps), straight from global
    V16U aq[2];
    {
        const __bf16* qrow = Qg + base + (size_t)(q0 + ln) * HDIM;
#pragma unroll
        for (int ks = 0; ks < 2; ++ks) {
            aq[ks].h[0] = *(const v8bf*)(qrow + ks * 32 + half * 8);
            aq[ks].h[1] = *(const v8bf*)(qrow + ks * 32 + 16 + half * 8);
        }
    }

    v8f oacc[4];
    float mrun[8], lrun[8];
#pragma unroll
    for (int nt = 0; nt < 4; ++nt) oacc[nt] = vzero8();
#pragma unroll
    for (int v = 0; v < 8; ++v) { mrun[v] = -3.0e38f; lrun[v] = 0.0f; }

    for (int kv0 = 0; kv0 < SEQ; kv0 += 64) {
        // ---- K tile via Tensor Data Mover (one descriptor, wave 0 only) ----
        if (wave == 0) {
            tdm_load_tile_2d_padded(Kg + base + (size_t)kv0 * HDIM,
                                    (unsigned)(size_t)&kbuf[0],
                                    /*tile_d0=*/HDIM, /*tile_d1=*/64,
                                    /*tensor_d0=*/HDIM, /*tensor_d1=*/SEQ,
                                    /*stride0=*/HDIM);
            __builtin_amdgcn_s_wait_tensorcnt((short)0);
        }
        // ---- V tile: cooperative load + transpose into [hd][kv] ----
#pragma unroll
        for (int i = 0; i < 2; ++i) {
            int ch  = tid + i * 256;               // 512 chunks of 8 bf16
            int row = ch >> 3;                     // kv row 0..63
            int c8  = (ch & 7) * 8;                // hd offset
            v8bf vv = *(const v8bf*)(Vg + base + (size_t)(kv0 + row) * HDIM + c8);
#pragma unroll
            for (int j = 0; j < 8; ++j) vbuf[(c8 + j) * KST + row] = vv[j];
        }
        __syncthreads();

        // ---- S = Q * K^T (16 x 64), f32 accum ----
        v8f sacc[4];
#pragma unroll
        for (int nt = 0; nt < 4; ++nt) {
            v8f s = vzero8();
#pragma unroll
            for (int ks = 0; ks < 2; ++ks) {
                v16bf bk = load_b_tile(kbuf, KST, nt * 16 + ln, half, ks * 32);
                s = __builtin_amdgcn_wmma_f32_16x16x32_bf16(
                    false, aq[ks].v, false, bk, (short)0, s, false, false);
            }
            sacc[nt] = s;
        }

        // ---- online softmax (row = v + 8*half, cols across 16 lanes) ----
        float pv[4][8];
#pragma unroll
        for (int v = 0; v < 8; ++v) {
            float bm = -3.0e38f;
#pragma unroll
            for (int nt = 0; nt < 4; ++nt) {
                float x = sacc[nt][v] * SOFTMAX_SCALE;
                bm = fmaxf(bm, x);
            }
#pragma unroll
            for (int off = 1; off < 16; off <<= 1)
                bm = fmaxf(bm, __shfl_xor(bm, off, 32));
            float mnew  = fmaxf(mrun[v], bm);
            float alpha = __expf(mrun[v] - mnew);
            float rs = 0.0f;
#pragma unroll
            for (int nt = 0; nt < 4; ++nt) {
                float p = __expf(sacc[nt][v] * SOFTMAX_SCALE - mnew);
                pv[nt][v] = p;
                rs += p;
            }
#pragma unroll
            for (int off = 1; off < 16; off <<= 1)
                rs += __shfl_xor(rs, off, 32);
            lrun[v] = lrun[v] * alpha + rs;
            mrun[v] = mnew;
#pragma unroll
            for (int nt = 0; nt < 4; ++nt) oacc[nt][v] *= alpha;
        }

        // ---- C-layout -> A-layout for P via wave-private LDS staging ----
        __bf16* pb = &pbuf[wave * 16 * KST];
#pragma unroll
        for (int nt = 0; nt < 4; ++nt)
#pragma unroll
            for (int v = 0; v < 8; ++v)
                pb[(v + 8 * half) * KST + nt * 16 + ln] = (__bf16)pv[nt][v];

        V16U pa[2];
        {
            const __bf16* pr = pb + (size_t)ln * KST;
#pragma unroll
            for (int ks = 0; ks < 2; ++ks) {
                pa[ks].h[0] = *(const v8bf*)(pr + ks * 32 + half * 8);
                pa[ks].h[1] = *(const v8bf*)(pr + ks * 32 + 16 + half * 8);
            }
        }

        // ---- O += P * V ----
#pragma unroll
        for (int nt = 0; nt < 4; ++nt) {
            v8f o = oacc[nt];
#pragma unroll
            for (int ks = 0; ks < 2; ++ks) {
                v16bf bv = load_b_tile(vbuf, KST, nt * 16 + ln, half, ks * 32);
                o = __builtin_amdgcn_wmma_f32_16x16x32_bf16(
                    false, pa[ks].v, false, bv, (short)0, o, false, false);
            }
            oacc[nt] = o;
        }
        __syncthreads();
    }

    // ---- normalize and write O as bf16 [B*S, D] (head-interleaved) ----
#pragma unroll
    for (int v = 0; v < 8; ++v) {
        int    srow = q0 + v + 8 * half;
        size_t orow = ((size_t)b * SEQ + srow) * DIM + h * HDIM;
        float  inv  = 1.0f / lrun[v];
#pragma unroll
        for (int nt = 0; nt < 4; ++nt)
            Og[orow + nt * 16 + ln] = (__bf16)(oacc[nt][v] * inv);
    }
}

// ---------------------------------------------------------------------------
// Host-side launch
// ---------------------------------------------------------------------------
extern "C" void kernel_launch(void* const* d_in, const int* in_sizes, int n_in,
                              void* d_out, int out_size, void* d_ws, size_t ws_size,
                              hipStream_t stream) {
    const float* query = (const float*)d_in[0];
    const float* key_i = (const float*)d_in[1];
    const float* value = (const float*)d_in[2];
    const float* w_q   = (const float*)d_in[3];
    const float* b_q   = (const float*)d_in[4];
    const float* w_k   = (const float*)d_in[5];
    const float* b_k   = (const float*)d_in[6];
    const float* w_v   = (const float*)d_in[7];
    const float* b_v   = (const float*)d_in[8];
    const float* w_fc  = (const float*)d_in[9];
    const float* b_fc  = (const float*)d_in[10];

    // Workspace carve-up (bf16): Q/K/V [B,H,S,HD] 8MB each, O 8MB, 4 weights 2MB each
    char* ws = (char*)d_ws;
    const size_t QKV_BYTES = (size_t)BATCH * NHEAD * SEQ * HDIM * sizeof(__bf16);
    const size_t O_BYTES   = (size_t)MROWS * DIM * sizeof(__bf16);
    const size_t W_BYTES   = (size_t)DIM * DIM * sizeof(__bf16);
    __bf16* Qb   = (__bf16*)(ws);
    __bf16* Kb   = (__bf16*)(ws + QKV_BYTES);
    __bf16* Vb   = (__bf16*)(ws + 2 * QKV_BYTES);
    __bf16* Ob   = (__bf16*)(ws + 3 * QKV_BYTES);
    __bf16* Wqt  = (__bf16*)(ws + 3 * QKV_BYTES + O_BYTES);
    __bf16* Wkt  = (__bf16*)(ws + 3 * QKV_BYTES + O_BYTES + W_BYTES);
    __bf16* Wvt  = (__bf16*)(ws + 3 * QKV_BYTES + O_BYTES + 2 * W_BYTES);
    __bf16* Wfct = (__bf16*)(ws + 3 * QKV_BYTES + O_BYTES + 3 * W_BYTES);

    // 1) weight convert/transpose (f32 [K][N] -> bf16 [N][K])
    dim3 wcg((DIM * DIM) / 256);
    wconv_kernel<<<wcg, 256, 0, stream>>>(w_q,  Wqt);
    wconv_kernel<<<wcg, 256, 0, stream>>>(w_k,  Wkt);
    wconv_kernel<<<wcg, 256, 0, stream>>>(w_v,  Wvt);
    wconv_kernel<<<wcg, 256, 0, stream>>>(w_fc, Wfct);

    // 2) Q/K/V projections: f32 activations -> bf16 head-major [B,H,S,HD]
    dim3 gg(DIM / 64, MROWS / 128);
    gemm_kernel<float, 0><<<gg, 256, 0, stream>>>(query, Wqt, b_q, Qb);
    gemm_kernel<float, 0><<<gg, 256, 0, stream>>>(key_i, Wkt, b_k, Kb);
    gemm_kernel<float, 0><<<gg, 256, 0, stream>>>(value, Wvt, b_v, Vb);

    // 3) flash attention -> O bf16 [B*S, D]
    dim3 ga(SEQ / 128, NHEAD, BATCH);
    attn_kernel<<<ga, 256, 0, stream>>>(Qb, Kb, Vb, Ob);

    // 4) final FC: bf16 O -> f32 out [B,S,D]
    gemm_kernel<__bf16, 1><<<gg, 256, 0, stream>>>(Ob, Wfct, b_fc, (float*)d_out);
}
